// InterObjectAttention_83296595738945
// MI455X (gfx1250) — matrix operand order repaired and checked by reading, not compiled, not run on hardware
//
#include <hip/hip_runtime.h>
#include <math.h>

typedef __attribute__((ext_vector_type(16))) _Float16 v16h;
typedef __attribute__((ext_vector_type(8)))  _Float16 v8h;
typedef __attribute__((ext_vector_type(8)))  float    v8f;

#define N_OBJ 256
#define P_PTS 256
#define D_DIM 16
#define K_NBR 4

// LDS layout (bytes). Pitches chosen for 16B alignment + bank spread.
#define QP  24             // sQ row pitch in halfs (48B, 16B aligned)
#define KP  264            // sK row pitch in halfs (528B, 16B aligned)
#define VP  24             // sV row pitch in halfs
#define SSP 264            // score slab row pitch in halfs (528B -> no bank clash)
#define SQ_OFF 0                       // 256*24*2  = 12288
#define SK_OFF 12288                   // 32*264*2  = 16896 (rows 16..31 zero pad)
#define SV_OFF 29184                   // 256*24*2  = 12288
#define SS_OFF 41472                   // 8 waves * 16*264 halfs = 67584
#define SN_OFF 109056                  // 4 ints
#define SMEM_BYTES 109072

// ---------------- Kernel 1: KNN (top-4) + positional encoding ----------------
__global__ void knn_pe_kernel(const float* __restrict__ positions,
                              const float* __restrict__ pos_W,
                              const float* __restrict__ pos_b,
                              int* __restrict__ nbr,
                              float* __restrict__ pe) {
  __shared__ float sp[N_OBJ][3];
  const int i = threadIdx.x;
  sp[i][0] = positions[i*3+0];
  sp[i][1] = positions[i*3+1];
  sp[i][2] = positions[i*3+2];
  __syncthreads();
  const float px = sp[i][0], py = sp[i][1], pz = sp[i][2];
  float bd0=INFINITY,bd1=INFINITY,bd2=INFINITY,bd3=INFINITY;
  int   bi0=0,bi1=0,bi2=0,bi3=0;
  for (int j = 0; j < N_OBJ; ++j) {
    if (j == i) continue;
    const float dx = px - sp[j][0], dy = py - sp[j][1], dz = pz - sp[j][2];
    const float d = sqrtf(dx*dx + dy*dy + dz*dz);
    // strict < keeps earlier index on ties (matches stable top_k over ascending j)
    if      (d < bd0) { bd3=bd2;bi3=bi2; bd2=bd1;bi2=bi1; bd1=bd0;bi1=bi0; bd0=d;bi0=j; }
    else if (d < bd1) { bd3=bd2;bi3=bi2; bd2=bd1;bi2=bi1; bd1=d;bi1=j; }
    else if (d < bd2) { bd3=bd2;bi3=bi2; bd2=d;bi2=j; }
    else if (d < bd3) { bd3=d;bi3=j; }
  }
  nbr[i*4+0]=bi0; nbr[i*4+1]=bi1; nbr[i*4+2]=bi2; nbr[i*4+3]=bi3;
  #pragma unroll
  for (int d = 0; d < D_DIM; ++d)
    pe[i*D_DIM+d] = px*pos_W[0*D_DIM+d] + py*pos_W[1*D_DIM+d] + pz*pos_W[2*D_DIM+d] + pos_b[d];
}

// ---------------- Kernel 2: attention + residual + LayerNorm ----------------
__global__ __launch_bounds__(256) void attn_kernel(
    const float* __restrict__ feats,
    const int*   __restrict__ nbr,
    const float* __restrict__ pe,
    const float* __restrict__ ln_gamma,
    const float* __restrict__ ln_beta,
    float* __restrict__ out) {
  extern __shared__ char smem[];
  _Float16* sQ = (_Float16*)(smem + SQ_OFF);   // [256][QP] Q * 1/sqrt(D), f16
  _Float16* sK = (_Float16*)(smem + SK_OFF);   // [32][KP]  Key^T; rows 16..31 = 0
  _Float16* sV = (_Float16*)(smem + SV_OFF);   // [256][VP] raw neighbor feats, f16
  _Float16* sS = (_Float16*)(smem + SS_OFF);   // per-wave 16xSSP score slab, f16
  int*    sNbr = (int*)(smem + SN_OFF);

  const int n    = blockIdx.x;
  const int tid  = threadIdx.x;
  const int lane = tid & 31;
  const int wave = tid >> 5;
  const int col  = lane & 15;     // N-column / d index within 16-lane half
  const int half = lane >> 4;     // lane half selects K-group in A fragments

  if (tid < K_NBR) sNbr[tid] = nbr[n*K_NBR + tid];

  // Zero pad rows 16..31 of sK once (K-dim padding lives in memory, not cndmasks)
  for (int t = tid; t < 16*KP; t += 256) sK[16*KP + t] = (_Float16)0.0f;

  // Build Q = (feats[n] + pe[n]) * 0.25 (1/sqrt(16) folded into A operand)
  {
    const float* fr  = feats + ((size_t)n * P_PTS + tid) * D_DIM;
    const float* per = pe + n * D_DIM;
    #pragma unroll
    for (int d = 0; d < D_DIM; ++d)
      sQ[tid*QP + d] = (_Float16)((fr[d] + per[d]) * 0.25f);
  }
  __syncthreads();

  v8f acc0, acc1;
  #pragma unroll
  for (int r = 0; r < 8; ++r) { acc0[r] = 0.0f; acc1[r] = 0.0f; }

  for (int k = 0; k < K_NBR; ++k) {
    const int nb = sNbr[k];
    if (k + 1 < K_NBR)  // warm L2/WGP$ for next neighbor tile
      __builtin_prefetch(feats + ((size_t)sNbr[k+1] * P_PTS + tid) * D_DIM, 0, 1);
    __syncthreads();    // previous k's sK/sV readers done
    {
      const float* fr  = feats + ((size_t)nb * P_PTS + tid) * D_DIM;
      const float* per = pe + nb * D_DIM;
      #pragma unroll
      for (int d = 0; d < D_DIM; ++d) {
        const float f = fr[d];
        sV[tid*VP + d] = (_Float16)f;            // V (no pe)
        sK[d*KP + tid] = (_Float16)(f + per[d]); // Key^T (rows 0..15 only)
      }
    }
    __syncthreads();

    #pragma unroll
    for (int tt = 0; tt < 2; ++tt) {
      const int pbase = tt*128 + wave*16;
      _Float16* slab = sS + wave * (16*SSP);

      // A fragment: rows p=pbase+col, K = d (zero-padded beyond 15)
      v16h afrag;
      #pragma unroll
      for (int e = 0; e < 16; ++e) afrag[e] = (_Float16)0.0f;
      {
        const v8h av = *(const v8h*)(sQ + (pbase + col)*QP + half*8);
        #pragma unroll
        for (int e = 0; e < 8; ++e) afrag[e] = av[e];
      }

      // ---- Pass A: scores strip (16 x 256) via 16 WMMAs ----
      #pragma unroll
      for (int j = 0; j < 16; ++j) {
        v16h bfrag;  // lane = q-column (col), elems = K rows (=d, table order; rows>=16 zero)
        {
          #pragma unroll
          for (int e = 0; e < 8; ++e) {
            bfrag[e]   = sK[(half*8 + e)*KP + j*16 + col];
            bfrag[8+e] = sK[(16 + half*8 + e)*KP + j*16 + col];
          }
        }
        v8f c;
        #pragma unroll
        for (int r = 0; r < 8; ++r) c[r] = 0.0f;
        c = __builtin_amdgcn_wmma_f32_16x16x32_f16(false, afrag, false, bfrag,
                                                   (short)0, c, false, false);
        #pragma unroll
        for (int r = 0; r < 8; ++r)
          slab[(r + half*8)*SSP + j*16 + col] = (_Float16)c[r];
      }

      // ---- Pass B: row softmax, normalized in place (lane pair per row) ----
      {
        _Float16* rowp = slab + col*SSP + half*128;
        float m = -INFINITY;
        for (int c2 = 0; c2 < 128; ++c2) m = fmaxf(m, (float)rowp[c2]);
        m = fmaxf(m, __shfl_xor(m, 16, 32));
        float sum = 0.0f;
        for (int c2 = 0; c2 < 128; ++c2) {
          const float e = __expf((float)rowp[c2] - m);
          sum += e;
          rowp[c2] = (_Float16)e;
        }
        sum += __shfl_xor(sum, 16, 32);
        const float inv = 1.0f / sum;
        for (int c2 = 0; c2 < 128; ++c2)
          rowp[c2] = (_Float16)((float)rowp[c2] * inv);
      }

      // ---- Pass C: attended += W @ V, 8 WMMAs (K=32 each) ----
      v8f* accp = tt ? &acc1 : &acc0;
      #pragma unroll
      for (int j2 = 0; j2 < 8; ++j2) {
        v16h wfrag;  // A: rows p, K = q
        {
          const v8h a0 = *(const v8h*)(slab + col*SSP + j2*32 + half*8);
          const v8h a1 = *(const v8h*)(slab + col*SSP + j2*32 + 16 + half*8);
          #pragma unroll
          for (int e = 0; e < 8; ++e) { wfrag[e] = a0[e]; wfrag[8+e] = a1[e]; }
        }
        v16h vfrag;  // B: lane = d column (col), elements = q rows (table order)
        {
          #pragma unroll
          for (int e = 0; e < 8; ++e) {
            vfrag[e]   = sV[(j2*32 + half*8 + e)*VP + col];
            vfrag[8+e] = sV[(j2*32 + 16 + half*8 + e)*VP + col];
          }
        }
        *accp = __builtin_amdgcn_wmma_f32_16x16x32_f16(false, wfrag, false, vfrag,
                                                       (short)0, *accp, false, false);
      }
    } // tt
  } // k

  // ---- Epilogue: residual + LayerNorm over D (16-lane reductions) ----
  const float g  = ln_gamma[col];
  const float bt = ln_beta[col];
  #pragma unroll
  for (int tt = 0; tt < 2; ++tt) {
    const v8f acc = tt ? acc1 : acc0;
    const int pbase = tt*128 + wave*16;
    #pragma unroll
    for (int r = 0; r < 8; ++r) {
      const int p = pbase + r + half*8;
      const size_t idx = (((size_t)n * P_PTS) + p) * D_DIM + col;
      const float u = feats[idx] + 0.2f * acc[r];
      float s = u;
      s += __shfl_xor(s, 1, 32); s += __shfl_xor(s, 2, 32);
      s += __shfl_xor(s, 4, 32); s += __shfl_xor(s, 8, 32);
      const float mean = s * (1.0f/16.0f);
      const float dv = u - mean;
      float v2 = dv*dv;
      v2 += __shfl_xor(v2, 1, 32); v2 += __shfl_xor(v2, 2, 32);
      v2 += __shfl_xor(v2, 4, 32); v2 += __shfl_xor(v2, 8, 32);
      const float var = v2 * (1.0f/16.0f);
      const float rs = rsqrtf(var + 1e-5f);
      out[idx] = dv * rs * g + bt;
    }
  }
}

extern "C" void kernel_launch(void* const* d_in, const int* in_sizes, int n_in,
                              void* d_out, int out_size, void* d_ws, size_t ws_size,
                              hipStream_t stream) {
  const float* feats     = (const float*)d_in[0];
  const float* positions = (const float*)d_in[1];
  const float* pos_W     = (const float*)d_in[2];
  const float* pos_b     = (const float*)d_in[3];
  const float* ln_gamma  = (const float*)d_in[4];
  const float* ln_beta   = (const float*)d_in[5];
  float* out = (float*)d_out;

  int*   nbr = (int*)d_ws;                                        // 256*4 ints
  float* pe  = (float*)((char*)d_ws + N_OBJ*K_NBR*sizeof(int));   // 256*16 floats

  knn_pe_kernel<<<1, N_OBJ, 0, stream>>>(positions, pos_W, pos_b, nbr, pe);
  attn_kernel<<<N_OBJ, 256, SMEM_BYTES, stream>>>(feats, nbr, pe, ln_gamma, ln_beta, out);
}
